// stack_CVAE_38886633898705
// MI455X (gfx1250) — matrix operand order repaired and checked by reading, not compiled, not run on hardware
//
#include <hip/hip_runtime.h>
#include <cstdint>

// ---------------------------------------------------------------------------
// Stack-GRU CVAE for MI455X (gfx1250), wave32 + bf16 WMMA (v_wmma_f32_16x16x32_bf16)
//  - decoder/encoder input-gate GEMMs hoisted out of the recurrence (big 2x2-blocked WMMA)
//  - serial steps use 1x1-tile WMMA GEMMs for max wave-level parallelism
// ---------------------------------------------------------------------------

typedef __attribute__((ext_vector_type(16))) __bf16 v16bf;
typedef __attribute__((ext_vector_type(8)))  float  v8f;

constexpr int B   = 128;
constexpr int T   = 128;
constexpr int H   = 768;
constexpr int L   = 128;
constexpr int P   = 3;
constexpr int SW  = 128;
constexpr int SD  = 64;
constexpr int VOUT = 100;
constexpr int H3  = 3 * H;       // 2304
constexpr int KXE = 800;         // H+P = 771 padded to multiple of 32
constexpr int KDX = 928;         // L+P+H = 899 padded to multiple of 32
constexpr int NWO = 112;         // VOUT padded to multiple of 16
constexpr int BT  = B * T;       // 16384

__device__ __forceinline__ float bf2f(unsigned short u) {
  return __uint_as_float(((unsigned int)u) << 16);
}
__device__ __forceinline__ unsigned short f2bf(float f) {
  unsigned int u = __float_as_uint(f);
  u += 0x7fffu + ((u >> 16) & 1u);   // round-to-nearest-even
  return (unsigned short)(u >> 16);
}

// Fragment loaders: both A (row-major MxK) and B (=W row-major NxK, used as K x N)
// reduce to two contiguous 16-byte row-segment loads per lane.
__device__ __forceinline__ v16bf load_fragA(const unsigned short* row, int k0, int half) {
  v16bf a;
  *((uint4*)&a)       = *(const uint4*)(row + k0 + 8 * half);        // K 0..7 / 8..15
  *(((uint4*)&a) + 1) = *(const uint4*)(row + k0 + 8 * half + 16);   // K 16..23 / 24..31
  return a;
}
__device__ __forceinline__ v16bf load_fragB(const unsigned short* row, int k0, int half) {
  v16bf b;
  *((uint4*)&b)       = *(const uint4*)(row + k0 + 8 * half);        // K 0..7 / 8..15
  *(((uint4*)&b) + 1) = *(const uint4*)(row + k0 + 8 * half + 16);   // K 16..23 / 24..31
  return b;
}

// -------------------- weight staging: fp32 -> bf16 with pad --------------------
__global__ void k_conv_pad(unsigned short* __restrict__ dst, const float* __restrict__ src,
                           int dstRows, int dstCols, int srcRows, int srcCols,
                           int srcStride, int srcColOff) {
  int idx = blockIdx.x * blockDim.x + threadIdx.x;
  int n = dstRows * dstCols;
  if (idx >= n) return;
  int r = idx / dstCols, c = idx % dstCols;
  float v = (r < srcRows && c < srcCols) ? src[(size_t)r * srcStride + srcColOff + c] : 0.0f;
  dst[idx] = f2bf(v);
}

// -------------------- activation builders --------------------
// rows ordered t-major: row = t*B + b (so per-step slices are contiguous)
__global__ void k_build_encx(unsigned short* __restrict__ dst, const int* __restrict__ inp,
                             const float* __restrict__ emb, const float* __restrict__ prop) {
  long long idx = (long long)blockIdx.x * blockDim.x + threadIdx.x;
  if (idx >= (long long)BT * KXE) return;
  int r = (int)(idx / KXE), c = (int)(idx % KXE);
  int t = r / B, b = r % B;
  float v;
  if (c < H)            v = emb[(size_t)inp[b * T + t] * H + c];
  else if (c < H + P)   v = prop[b * P + (c - H)];
  else                  v = 0.0f;
  dst[idx] = f2bf(v);
}

__global__ void k_build_decx(unsigned short* __restrict__ dst, const int* __restrict__ inp,
                             const float* __restrict__ emb, const float* __restrict__ prop,
                             const float* __restrict__ z) {
  long long idx = (long long)blockIdx.x * blockDim.x + threadIdx.x;
  if (idx >= (long long)BT * KDX) return;
  int r = (int)(idx / KDX), c = (int)(idx % KDX);
  int t = r / B, b = r % B;
  float v;
  if (c < L)               v = z[b * L + c];
  else if (c < L + P)      v = prop[b * P + (c - L)];
  else if (c < L + P + H)  v = emb[(size_t)inp[b * T + t] * H + (c - L - P)];
  else                     v = 0.0f;
  dst[idx] = f2bf(v);
}

// -------------------- state init --------------------
__global__ void k_copy_h(const float* __restrict__ src, float* __restrict__ h,
                         unsigned short* __restrict__ hbf) {
  int idx = blockIdx.x * blockDim.x + threadIdx.x;
  if (idx >= B * H) return;
  float v = src[idx];
  h[idx] = v; hbf[idx] = f2bf(v);
}

__global__ void k_copy_stack(const float* __restrict__ src, float* __restrict__ dst) {
  int idx = blockIdx.x * blockDim.x + threadIdx.x;
  if (idx >= B * SD * SW) return;
  dst[idx] = src[idx];
}

// -------------------- 1x1-tile bf16 WMMA GEMM: out = A(MxK) @ W(NxK)^T --------------------
// Used for the latency-bound per-step GEMMs (M=128): maximize wave count.
// grid = (N/128, M/16), block = 256 (8 waves, one 16x16 tile each)
__global__ void k_gemm_bf16(const unsigned short* __restrict__ A, int lda,
                            const unsigned short* __restrict__ W, int ldw,
                            int K,
                            const float* __restrict__ bias,
                            const unsigned short* __restrict__ C0, int ldc0,
                            float* __restrict__ outF,
                            unsigned short* __restrict__ outB, int ldo) {
  int lane = threadIdx.x & 31;
  int wave = threadIdx.x >> 5;
  int half = lane >> 4;
  int l15  = lane & 15;
  int tileM = blockIdx.y * 16;
  int tileN = blockIdx.x * 128 + wave * 16;

  const unsigned short* arow = A + (size_t)(tileM + l15) * lda;
  const unsigned short* wrow = W + (size_t)(tileN + l15) * ldw;

  v8f acc = {};
  for (int k0 = 0; k0 < K; k0 += 32) {
    v16bf a = load_fragA(arow, k0, half);
    v16bf b = load_fragB(wrow, k0, half);
    acc = __builtin_amdgcn_wmma_f32_16x16x32_bf16(false, a, false, b, (short)0, acc, false, false);
  }

  int col = tileN + l15;
  float bv = bias ? bias[col] : 0.0f;
#pragma unroll
  for (int r = 0; r < 8; ++r) {
    int row = tileM + r + 8 * half;          // C/D layout: VGPR r -> M = r + 8*(lane>>4)
    float v = acc[r] + bv;
    if (C0) v += bf2f(C0[(size_t)row * ldc0 + col]);
    if (outF) outF[(size_t)row * ldo + col] = v;
    else      outB[(size_t)row * ldo + col] = f2bf(v);
  }
}

// -------------------- 2x2 register-blocked bf16 WMMA GEMM (throughput path) ---------------
// outB(bf16) = A(MxK) @ W(NxK)^T + bias. Each wave computes a 32x32 tile:
// A fragments reused across 2 N-tiles, B fragments across 2 M-tiles -> 2 b128 loads/WMMA.
// grid = (N/256, M/32), block = 256 (8 waves * 32 cols = 256 cols/block)
__global__ void k_gemm_bf16_2x2(const unsigned short* __restrict__ A, int lda,
                                const unsigned short* __restrict__ W, int ldw,
                                int K,
                                const float* __restrict__ bias,
                                unsigned short* __restrict__ outB, int ldo) {
  int lane = threadIdx.x & 31;
  int wave = threadIdx.x >> 5;
  int half = lane >> 4;
  int l15  = lane & 15;
  int tileM = blockIdx.y * 32;
  int tileN = blockIdx.x * 256 + wave * 32;

  const unsigned short* arow0 = A + (size_t)(tileM + l15) * lda;
  const unsigned short* arow1 = arow0 + (size_t)16 * lda;
  const unsigned short* wrow0 = W + (size_t)(tileN + l15) * ldw;
  const unsigned short* wrow1 = wrow0 + (size_t)16 * ldw;

  v8f acc00 = {}, acc01 = {}, acc10 = {}, acc11 = {};
  for (int k0 = 0; k0 < K; k0 += 32) {
    v16bf a0 = load_fragA(arow0, k0, half);
    v16bf a1 = load_fragA(arow1, k0, half);
    v16bf b0 = load_fragB(wrow0, k0, half);
    v16bf b1 = load_fragB(wrow1, k0, half);
    acc00 = __builtin_amdgcn_wmma_f32_16x16x32_bf16(false, a0, false, b0, (short)0, acc00, false, false);
    acc01 = __builtin_amdgcn_wmma_f32_16x16x32_bf16(false, a0, false, b1, (short)0, acc01, false, false);
    acc10 = __builtin_amdgcn_wmma_f32_16x16x32_bf16(false, a1, false, b0, (short)0, acc10, false, false);
    acc11 = __builtin_amdgcn_wmma_f32_16x16x32_bf16(false, a1, false, b1, (short)0, acc11, false, false);
  }

  int col0 = tileN + l15, col1 = col0 + 16;
  float bv0 = bias ? bias[col0] : 0.0f;
  float bv1 = bias ? bias[col1] : 0.0f;
#pragma unroll
  for (int r = 0; r < 8; ++r) {
    int row0 = tileM + r + 8 * half;
    int row1 = row0 + 16;
    outB[(size_t)row0 * ldo + col0] = f2bf(acc00[r] + bv0);
    outB[(size_t)row0 * ldo + col1] = f2bf(acc01[r] + bv1);
    outB[(size_t)row1 * ldo + col0] = f2bf(acc10[r] + bv0);
    outB[(size_t)row1 * ldo + col1] = f2bf(acc11[r] + bv1);
  }
}

// -------------------- final projection: d_out(B,T,VOUT) = dec_out @ W_out^T + b --------------------
// block = 224 (7 waves covering N=112 padded cols), grid = BT/16
__global__ void k_gemm_final(const unsigned short* __restrict__ A,   // BT x H bf16 (t-major rows)
                             const unsigned short* __restrict__ W,   // NWO x H bf16 (rows >=100 zero)
                             const float* __restrict__ bias,
                             float* __restrict__ out) {
  int lane = threadIdx.x & 31;
  int wave = threadIdx.x >> 5;   // 0..6
  int half = lane >> 4;
  int l15  = lane & 15;
  int tileM = blockIdx.x * 16;
  int tileN = wave * 16;

  const unsigned short* arow = A + (size_t)(tileM + l15) * H;
  const unsigned short* wrow = W + (size_t)(tileN + l15) * H;

  v8f acc = {};
  for (int k0 = 0; k0 < H; k0 += 32) {
    v16bf a = load_fragA(arow, k0, half);
    v16bf b = load_fragB(wrow, k0, half);
    acc = __builtin_amdgcn_wmma_f32_16x16x32_bf16(false, a, false, b, (short)0, acc, false, false);
  }

  int col = tileN + l15;
  if (col < VOUT) {
    float bv = bias[col];
#pragma unroll
    for (int r = 0; r < 8; ++r) {
      int row = tileM + r + 8 * half;   // row = t*B + b
      int t = row >> 7, bb = row & (B - 1);
      out[(size_t)bb * (T * VOUT) + (size_t)t * VOUT + col] = acc[r] + bv;
    }
  }
}

// -------------------- per-step controller: ctrl softmax + s_in = tanh(h@Ws^T+bs) ------------
// grid = B, block = SW (=128)
__global__ void k_ctrl_sin(const float* __restrict__ h,
                           const float* __restrict__ Wc, const float* __restrict__ bc,
                           const float* __restrict__ Ws, const float* __restrict__ bs,
                           float* __restrict__ ctrl, float* __restrict__ s_in) {
  __shared__ float hs[H];
  int b = blockIdx.x, j = threadIdx.x;
  for (int i = j; i < H; i += SW) hs[i] = h[(size_t)b * H + i];
  __syncthreads();
  const float* wr = Ws + (size_t)j * H;
  float acc = bs[j];
  for (int k = 0; k < H; ++k) acc += hs[k] * wr[k];
  s_in[b * SW + j] = tanhf(acc);
  if (j == 0) {
    float c0 = bc[0], c1 = bc[1], c2 = bc[2];
    for (int k = 0; k < H; ++k) {
      float hv = hs[k];
      c0 += hv * Wc[k]; c1 += hv * Wc[H + k]; c2 += hv * Wc[2 * H + k];
    }
    float m = fmaxf(c0, fmaxf(c1, c2));
    float e0 = expf(c0 - m), e1 = expf(c1 - m), e2 = expf(c2 - m);
    float s = e0 + e1 + e2;
    ctrl[b * 3 + 0] = e0 / s; ctrl[b * 3 + 1] = e1 / s; ctrl[b * 3 + 2] = e2 / s;
  }
}

// -------------------- stack update (ping-pong) + capture new top in bf16 --------------------
__global__ void k_stack(const float* __restrict__ src, float* __restrict__ dst,
                        const float* __restrict__ ctrl, const float* __restrict__ s_in,
                        unsigned short* __restrict__ s_top) {
  int idx = blockIdx.x * blockDim.x + threadIdx.x;
  if (idx >= B * SD * SW) return;
  int b = idx / (SD * SW);
  int rem = idx - b * (SD * SW);
  int d = rem / SW, w = rem - d * SW;
  float push = ctrl[b * 3 + 0], pop = ctrl[b * 3 + 1], noop = ctrl[b * 3 + 2];
  float st   = src[idx];
  float up   = (d == 0)      ? s_in[b * SW + w] : src[idx - SW];
  float down = (d == SD - 1) ? 0.0f             : src[idx + SW];
  float v = noop * st + push * up + pop * down;
  dst[idx] = v;
  if (d == 0) s_top[b * SW + w] = f2bf(v);
}

// -------------------- fused GRU pointwise (encoder: gi fp32) --------------------
__global__ void k_gru(const float* __restrict__ gi, const float* __restrict__ gh,
                      float* __restrict__ h, unsigned short* __restrict__ hbf) {
  int idx = blockIdx.x * blockDim.x + threadIdx.x;
  if (idx >= B * H) return;
  int b = idx / H, j = idx - b * H;
  size_t base = (size_t)b * H3;
  float ir = gi[base + j], iz = gi[base + H + j], inn = gi[base + 2 * H + j];
  float hr = gh[base + j], hz = gh[base + H + j], hn  = gh[base + 2 * H + j];
  float r = 1.0f / (1.0f + expf(-(ir + hr)));
  float z = 1.0f / (1.0f + expf(-(iz + hz)));
  float n = tanhf(inn + r * hn);
  float hv = h[idx];
  float hnew = (1.0f - z) * n + z * hv;
  h[idx] = hnew; hbf[idx] = f2bf(hnew);
}

// -------------------- fused GRU pointwise (decoder: gi precomputed bf16, emit dec_out) ------
__global__ void k_gru_dec(const unsigned short* __restrict__ gi, const float* __restrict__ gh,
                          float* __restrict__ h, unsigned short* __restrict__ hbf,
                          unsigned short* __restrict__ dec_out) {
  int idx = blockIdx.x * blockDim.x + threadIdx.x;
  if (idx >= B * H) return;
  int b = idx / H, j = idx - b * H;
  size_t base = (size_t)b * H3;
  float ir = bf2f(gi[base + j]), iz = bf2f(gi[base + H + j]), inn = bf2f(gi[base + 2 * H + j]);
  float hr = gh[base + j], hz = gh[base + H + j], hn = gh[base + 2 * H + j];
  float r = 1.0f / (1.0f + expf(-(ir + hr)));
  float z = 1.0f / (1.0f + expf(-(iz + hz)));
  float n = tanhf(inn + r * hn);
  float hv = h[idx];
  float hnew = (1.0f - z) * n + z * hv;
  h[idx] = hnew; hbf[idx] = f2bf(hnew);
  dec_out[idx] = f2bf(hnew);           // caller pre-offsets dec_out by t*B*H
}

// -------------------- latent: z = eps*exp(0.5*logv) + mean --------------------
// grid = B, block = L
__global__ void k_latent(const float* __restrict__ h,
                         const float* __restrict__ Wm, const float* __restrict__ bm,
                         const float* __restrict__ Wl, const float* __restrict__ bl,
                         const float* __restrict__ eps, float* __restrict__ z) {
  __shared__ float hs[H];
  int b = blockIdx.x, l = threadIdx.x;
  for (int i = l; i < H; i += L) hs[i] = h[(size_t)b * H + i];
  __syncthreads();
  const float* wm = Wm + (size_t)l * H;
  const float* wl = Wl + (size_t)l * H;
  float m = bm[l], v = bl[l];
  for (int k = 0; k < H; ++k) { float hv = hs[k]; m += hv * wm[k]; v += hv * wl[k]; }
  z[b * L + l] = eps[b * L + l] * expf(0.5f * v) + m;
}

// ===========================================================================
extern "C" void kernel_launch(void* const* d_in, const int* in_sizes, int n_in,
                              void* d_out, int out_size, void* d_ws, size_t ws_size,
                              hipStream_t stream) {
  (void)in_sizes; (void)n_in; (void)out_size; (void)ws_size;

  const int*   inp       = (const int*)  d_in[0];
  const float* prop      = (const float*)d_in[2];
  const float* enc_h0    = (const float*)d_in[3];
  const float* dec_h0    = (const float*)d_in[4];
  const float* enc_st0   = (const float*)d_in[5];
  const float* eps       = (const float*)d_in[6];
  const float* emb       = (const float*)d_in[7];
  const float* Wc        = (const float*)d_in[8];
  const float* bc        = (const float*)d_in[9];
  const float* Ws        = (const float*)d_in[10];
  const float* bs        = (const float*)d_in[11];
  const float* enc_Wi    = (const float*)d_in[12];
  const float* enc_Wh    = (const float*)d_in[13];
  const float* enc_bi    = (const float*)d_in[14];
  const float* enc_bh    = (const float*)d_in[15];
  const float* dec_Wi    = (const float*)d_in[16];
  const float* dec_Wh    = (const float*)d_in[17];
  const float* dec_bi    = (const float*)d_in[18];
  const float* dec_bh    = (const float*)d_in[19];
  const float* W_mean    = (const float*)d_in[20];
  const float* b_mean    = (const float*)d_in[21];
  const float* W_logv    = (const float*)d_in[22];
  const float* b_logv    = (const float*)d_in[23];
  const float* W_out     = (const float*)d_in[24];
  const float* b_out     = (const float*)d_in[25];
  float* out = (float*)d_out;

  // ---- workspace carve-out (~262 MB total) ----
  char* base = (char*)d_ws;
  size_t cur = 0;
  auto alloc = [&](size_t bytes) -> void* {
    void* p = base + cur;
    cur += (bytes + 255) & ~(size_t)255;
    return p;
  };
  unsigned short* wEncWiX = (unsigned short*)alloc((size_t)H3 * KXE * 2);
  unsigned short* wEncWiS = (unsigned short*)alloc((size_t)H3 * SW  * 2);
  unsigned short* wEncWh  = (unsigned short*)alloc((size_t)H3 * H   * 2);
  unsigned short* wDecWi  = (unsigned short*)alloc((size_t)H3 * KDX * 2);
  unsigned short* wDecWh  = (unsigned short*)alloc((size_t)H3 * H   * 2);
  unsigned short* wWout   = (unsigned short*)alloc((size_t)NWO * H  * 2);
  unsigned short* encX    = (unsigned short*)alloc((size_t)BT * KXE * 2);
  unsigned short* decX    = (unsigned short*)alloc((size_t)BT * KDX * 2);
  unsigned short* encGiX  = (unsigned short*)alloc((size_t)BT * H3  * 2);  // precomputed x-part gi (+bi)
  unsigned short* decGi   = (unsigned short*)alloc((size_t)BT * H3  * 2);  // precomputed full dec gi (+bi)
  unsigned short* decOut  = (unsigned short*)alloc((size_t)BT * H   * 2);
  float* stackA = (float*)alloc((size_t)B * SD * SW * 4);
  float* stackB = (float*)alloc((size_t)B * SD * SW * 4);
  float* hF     = (float*)alloc((size_t)B * H * 4);
  unsigned short* hBF = (unsigned short*)alloc((size_t)B * H * 2);
  float* giBuf  = (float*)alloc((size_t)B * H3 * 4);
  float* ghBuf  = (float*)alloc((size_t)B * H3 * 4);
  float* ctrl   = (float*)alloc((size_t)B * 3 * 4);
  float* sIn    = (float*)alloc((size_t)B * SW * 4);
  unsigned short* sTop = (unsigned short*)alloc((size_t)B * SW * 2);
  float* zBuf   = (float*)alloc((size_t)B * L * 4);

  auto g1 = [](long long n) { return dim3((unsigned)((n + 255) / 256)); };

  // ---- stage weights to bf16 (K padded with zeros) ----
  k_conv_pad<<<g1((long long)H3 * KXE), 256, 0, stream>>>(wEncWiX, enc_Wi, H3, KXE, H3, H + P, H + P + SW, 0);
  k_conv_pad<<<g1((long long)H3 * SW),  256, 0, stream>>>(wEncWiS, enc_Wi, H3, SW,  H3, SW,    H + P + SW, H + P);
  k_conv_pad<<<g1((long long)H3 * H),   256, 0, stream>>>(wEncWh,  enc_Wh, H3, H,   H3, H,     H, 0);
  k_conv_pad<<<g1((long long)H3 * KDX), 256, 0, stream>>>(wDecWi,  dec_Wi, H3, KDX, H3, L + P + H, L + P + H, 0);
  k_conv_pad<<<g1((long long)H3 * H),   256, 0, stream>>>(wDecWh,  dec_Wh, H3, H,   H3, H,     H, 0);
  k_conv_pad<<<g1((long long)NWO * H),  256, 0, stream>>>(wWout,   W_out,  NWO, H,  VOUT, H,   H, 0);

  // ---- encoder x-part gi for all (t,b): encGiX = encX @ WiX^T + enc_bi (throughput GEMM) ----
  k_build_encx<<<g1((long long)BT * KXE), 256, 0, stream>>>(encX, inp, emb, prop);
  k_gemm_bf16_2x2<<<dim3(H3 / 256, BT / 32), 256, 0, stream>>>(
      encX, KXE, wEncWiX, KXE, KXE, enc_bi, encGiX, H3);

  // ---- encoder recurrence ----
  k_copy_h<<<g1(B * H), 256, 0, stream>>>(enc_h0, hF, hBF);
  k_copy_stack<<<g1((long long)B * SD * SW), 256, 0, stream>>>(enc_st0, stackA);
  for (int t = 0; t < T; ++t) {
    const float* sSrc = (t & 1) ? stackB : stackA;
    float*       sDst = (t & 1) ? stackA : stackB;
    k_ctrl_sin<<<B, SW, 0, stream>>>(hF, Wc, bc, Ws, bs, ctrl, sIn);
    k_stack<<<g1((long long)B * SD * SW), 256, 0, stream>>>(sSrc, sDst, ctrl, sIn, sTop);
    // gi = precomputed x-part + stack-top part
    k_gemm_bf16<<<dim3(H3 / 128, B / 16), 256, 0, stream>>>(
        sTop, SW, wEncWiS, SW, SW, nullptr, encGiX + (size_t)t * B * H3, H3, giBuf, nullptr, H3);
    // gh = h @ enc_Wh^T + enc_bh
    k_gemm_bf16<<<dim3(H3 / 128, B / 16), 256, 0, stream>>>(
        hBF, H, wEncWh, H, H, enc_bh, nullptr, 0, ghBuf, nullptr, H3);
    k_gru<<<g1(B * H), 256, 0, stream>>>(giBuf, ghBuf, hF, hBF);
  }

  // ---- latent + decoder gi precompute (throughput GEMM) ----
  k_latent<<<B, L, 0, stream>>>(hF, W_mean, b_mean, W_logv, b_logv, eps, zBuf);
  k_build_decx<<<g1((long long)BT * KDX), 256, 0, stream>>>(decX, inp, emb, prop, zBuf);
  k_gemm_bf16_2x2<<<dim3(H3 / 256, BT / 32), 256, 0, stream>>>(
      decX, KDX, wDecWi, KDX, KDX, dec_bi, decGi, H3);

  // ---- decoder recurrence ----
  k_copy_h<<<g1(B * H), 256, 0, stream>>>(dec_h0, hF, hBF);
  for (int t = 0; t < T; ++t) {
    k_gemm_bf16<<<dim3(H3 / 128, B / 16), 256, 0, stream>>>(
        hBF, H, wDecWh, H, H, dec_bh, nullptr, 0, ghBuf, nullptr, H3);
    k_gru_dec<<<g1(B * H), 256, 0, stream>>>(
        decGi + (size_t)t * B * H3, ghBuf, hF, hBF, decOut + (size_t)t * B * H);
  }

  // ---- output projection: (B,T,VOUT) ----
  k_gemm_final<<<BT / 16, 224, 0, stream>>>(decOut, wWout, b_out, out);
}
